// CrossAttention_88081189306550
// MI455X (gfx1250) — compile-verified, matrix operand order checked by reading, not run on hardware
//
#include <hip/hip_runtime.h>

#define B_  16
#define NI_ 2048
#define NA_ 2048
#define D_  256

typedef __bf16 bf16;
typedef __attribute__((ext_vector_type(16))) __bf16 v16bf;
typedef __attribute__((ext_vector_type(8)))  __bf16 v8bf;
typedef __attribute__((ext_vector_type(8)))  float  v8f;
typedef __attribute__((ext_vector_type(4)))  float  v4f;
typedef __attribute__((ext_vector_type(4)))  unsigned int v4u;
typedef __attribute__((ext_vector_type(8)))  int v8i;
typedef __attribute__((ext_vector_type(4)))  int v4i;

// LDS tile geometry (strides padded so 16-lane b128 row accesses hit distinct banks)
#define KT  128   // keys per tile
#define KS  264   // K tile row stride (bf16): 528B -> bank shift 4 (TDM pad reproduces this)
#define VS  136   // V^T row stride: 272B -> bank shift 4
#define PS  136   // P row stride

__device__ __forceinline__ v16bf cat8(v8bf a, v8bf b) {
  return __builtin_shufflevector(a, b, 0,1,2,3,4,5,6,7,8,9,10,11,12,13,14,15);
}

__device__ __forceinline__ v8bf cvt8(const float* p) {
  v4f f0 = *(const v4f*)p;
  v4f f1 = *(const v4f*)(p + 4);
  v8bf h;
#pragma unroll
  for (int i = 0; i < 4; ++i) { h[i] = (bf16)f0[i]; h[i + 4] = (bf16)f1[i]; }
  return h;
}

// WMMA 16x32 bf16 per-lane operand: K = base..base+7 and base+16..base+23
__device__ __forceinline__ v16bf ld16(const bf16* p) {
  v8bf a = *(const v8bf*)p;
  v8bf b = *(const v8bf*)(p + 16);
  return cat8(a, b);
}

// ------------------------------------------------------------ W -> W^T bf16
__global__ void wt_kernel(const float* __restrict__ W, bf16* __restrict__ WT) {
  int idx = blockIdx.x * 256 + threadIdx.x;     // 65536 elems, grid 256
  int d = idx >> 8, e = idx & 255;
  WT[e * 256 + d] = (bf16)W[d * 256 + e];
}

// ------------------------------------------------------------ attendee f32 -> bf16
__global__ void conv_kernel(const float* __restrict__ src, bf16* __restrict__ dst) {
  size_t i = ((size_t)blockIdx.x * 256 + threadIdx.x) * 8;
  *(v8bf*)(dst + i) = cvt8(src + i);
}

// ------------------------------------------------------------ Q = X @ W (bf16 out)
__global__ void qproj_kernel(const float* __restrict__ X,
                             const bf16* __restrict__ WT,
                             bf16* __restrict__ Qb) {
  const int lane = threadIdx.x & 31;
  const int wave = threadIdx.x >> 5;
  const int rb   = (blockIdx.x * 8 + wave) * 16;
  const int n    = lane & 15;
  const int kb   = (lane < 16) ? 0 : 8;
  const int half = (lane < 16) ? 0 : 8;

  v16bf a[8];
  const float* xrow = X + (size_t)(rb + n) * D_;
#pragma unroll
  for (int ko = 0; ko < 8; ++ko) {
    const float* p = xrow + ko * 32 + kb;
    a[ko] = cat8(cvt8(p), cvt8(p + 16));
  }

#pragma unroll
  for (int ct = 0; ct < 16; ++ct) {
    v8f acc = {0.f,0.f,0.f,0.f,0.f,0.f,0.f,0.f};
#pragma unroll
    for (int h = 0; h < 2; ++h) {          // batch 4 B-operands, then 4 WMMAs
      v16bf bm[4];
#pragma unroll
      for (int k4 = 0; k4 < 4; ++k4)
        bm[k4] = ld16(WT + (ct * 16 + n) * D_ + (h * 4 + k4) * 32 + kb);
#pragma unroll
      for (int k4 = 0; k4 < 4; ++k4)
        acc = __builtin_amdgcn_wmma_f32_16x16x32_bf16(false, a[h * 4 + k4], false,
                                                      bm[k4], (short)0, acc, false, false);
    }
#pragma unroll
    for (int r = 0; r < 8; ++r) {
      int row = rb + r + half;
      Qb[(size_t)row * D_ + ct * 16 + n] = (bf16)acc[r];
    }
  }
}

// ------------------------------------------------------------ flash attention
// Grid: B * (NI/128) = 256 blocks of 256 threads (8 waves); wave owns 16 queries.
__global__ void flash_kernel(const bf16* __restrict__ Qb,
                             const bf16* __restrict__ Ab,
                             float* __restrict__ out) {
  extern __shared__ char smem[];
  bf16* Klds  = (bf16*)smem;                                  // [KT][KS], LDS offset 0
  bf16* VTlds = (bf16*)(smem + KT * KS * 2);                  // [D_][VS]
  bf16* Plds  = (bf16*)(smem + KT * KS * 2 + D_ * VS * 2);    // 8 x [16][PS]

  const int tid  = threadIdx.x;
  const int lane = tid & 31;
  const int wave = tid >> 5;
  const int b    = blockIdx.x >> 4;
  const int qt   = blockIdx.x & 15;
  const int n    = lane & 15;
  const int kb   = (lane < 16) ? 0 : 8;
  const int half = (lane < 16) ? 0 : 8;

  bf16* Pw = Plds + wave * 16 * PS;

  const int qrow = qt * 128 + wave * 16 + n;
  const bf16* qp = Qb + ((size_t)b * NI_ + qrow) * D_;
  v16bf qa[8];
#pragma unroll
  for (int ko = 0; ko < 8; ++ko) qa[ko] = ld16(qp + ko * 32 + kb);

  v8f O[16];
#pragma unroll
  for (int i = 0; i < 16; ++i) O[i] = (v8f){0.f,0.f,0.f,0.f,0.f,0.f,0.f,0.f};
  float m[8], l[8];
#pragma unroll
  for (int r = 0; r < 8; ++r) { m[r] = -3.0e38f; l[r] = 0.f; }

  const bf16* abase = Ab + (size_t)b * NA_ * D_;

  for (int kt = 0; kt < NA_ / KT; ++kt) {
    __syncthreads();                       // previous tile fully consumed
    if (wave == 0) {
      // TDM: DMA 128x256 bf16 tile -> Klds with 4-DWORD pad per 512B row (stride 264)
      unsigned long long ga =
          (unsigned long long)(uintptr_t)(abase + (size_t)kt * KT * D_);
      v4u g0 = { 1u,                                  // count=1
                 0u,                                  // lds_addr = 0 (Klds)
                 (unsigned int)(ga & 0xffffffffull),
                 (unsigned int)((ga >> 32) & 0x01ffffffull) | 0x80000000u }; // type=2
      v8i g1 = { 0x07910000,                          // data_size=2B, pad: int=6 amt=3
                 0x01000000,                          // tensor_dim0 = 256 (<<16)
                 (int)0x80000000,                     // tensor_dim1 = 32768 (lo16<<16)
                 0x01000000,                          // tile_dim0 = 256 (<<16)
                 128,                                 // tile_dim1 = 128
                 256,                                 // tensor_dim0_stride = 256
                 0, 0 };
      v4i z4 = {0, 0, 0, 0};
      v8i z8 = {0, 0, 0, 0, 0, 0, 0, 0};
      __builtin_amdgcn_tensor_load_to_lds(g0, g1, z4, z4, z8, 0);
      __builtin_amdgcn_s_wait_tensorcnt(0);
    }
    __syncthreads();                       // K tile visible to all waves

    // Build V^T from K tile (LDS -> LDS transpose through registers)
    for (int rr = 0; rr < 16; ++rr) {
      int chunk = rr * 256 + tid;          // 4096 chunks of 8 elems
      int key   = chunk >> 5;
      int dpos  = (chunk & 31) << 3;
      v8bf h = *(const v8bf*)(Klds + key * KS + dpos);
#pragma unroll
      for (int i = 0; i < 8; ++i) VTlds[(dpos + i) * VS + key] = h[i];
    }
    __syncthreads();

    // S = Q K^T : 8 column tiles x 8 k-steps (batched loads then WMMA bursts)
    v8f s[8];
#pragma unroll
    for (int ct = 0; ct < 8; ++ct) {
      v8f acc = {0.f,0.f,0.f,0.f,0.f,0.f,0.f,0.f};
#pragma unroll
      for (int h = 0; h < 2; ++h) {
        v16bf bm[4];
#pragma unroll
        for (int k4 = 0; k4 < 4; ++k4)
          bm[k4] = ld16(Klds + (ct * 16 + n) * KS + (h * 4 + k4) * 32 + kb);
#pragma unroll
        for (int k4 = 0; k4 < 4; ++k4)
          acc = __builtin_amdgcn_wmma_f32_16x16x32_bf16(false, qa[h * 4 + k4], false,
                                                        bm[k4], (short)0, acc, false, false);
      }
      s[ct] = acc;
    }

    // Online softmax (rows live per (VGPR r, lane-half))
    float t[8];
#pragma unroll
    for (int r = 0; r < 8; ++r) t[r] = s[0][r];
#pragma unroll
    for (int ct = 1; ct < 8; ++ct)
#pragma unroll
      for (int r = 0; r < 8; ++r) t[r] = fmaxf(t[r], s[ct][r]);
#pragma unroll
    for (int mask = 1; mask < 16; mask <<= 1)
#pragma unroll
      for (int r = 0; r < 8; ++r) t[r] = fmaxf(t[r], __shfl_xor(t[r], mask, 32));

    float sc[8];
#pragma unroll
    for (int r = 0; r < 8; ++r) {
      float mn = fmaxf(m[r], t[r]);
      sc[r] = __expf(m[r] - mn);
      m[r] = mn;
    }

    float rs[8];
#pragma unroll
    for (int r = 0; r < 8; ++r) rs[r] = 0.f;
#pragma unroll
    for (int ct = 0; ct < 8; ++ct)
#pragma unroll
      for (int r = 0; r < 8; ++r) {
        float p = __expf(s[ct][r] - m[r]);
        rs[r] += p;
        Pw[(r + half) * PS + ct * 16 + n] = (bf16)p;   // C-layout -> LDS
      }
#pragma unroll
    for (int mask = 1; mask < 16; mask <<= 1)
#pragma unroll
      for (int r = 0; r < 8; ++r) rs[r] += __shfl_xor(rs[r], mask, 32);
#pragma unroll
    for (int r = 0; r < 8; ++r) l[r] = l[r] * sc[r] + rs[r];
#pragma unroll
    for (int dt = 0; dt < 16; ++dt)
#pragma unroll
      for (int r = 0; r < 8; ++r) O[dt][r] *= sc[r];

    asm volatile("s_wait_dscnt 0" ::: "memory");   // our P stores -> our reads

    v16bf pa[4];
#pragma unroll
    for (int ks = 0; ks < 4; ++ks) pa[ks] = ld16(Pw + n * PS + ks * 32 + kb);

    // O += P V : 16 d-tiles x 4 k-steps
#pragma unroll
    for (int dt = 0; dt < 16; ++dt) {
      v16bf bm[4];
#pragma unroll
      for (int ks = 0; ks < 4; ++ks)
        bm[ks] = ld16(VTlds + (dt * 16 + n) * VS + ks * 32 + kb);
      v8f acc = O[dt];
#pragma unroll
      for (int ks = 0; ks < 4; ++ks)
        acc = __builtin_amdgcn_wmma_f32_16x16x32_bf16(false, pa[ks], false,
                                                      bm[ks], (short)0, acc, false, false);
      O[dt] = acc;
    }
  }

  // Epilogue: normalize by row sum, store f32
#pragma unroll
  for (int dt = 0; dt < 16; ++dt)
#pragma unroll
    for (int r = 0; r < 8; ++r) {
      int row = qt * 128 + wave * 16 + r + half;
      out[((size_t)b * NI_ + row) * D_ + dt * 16 + n] = O[dt][r] / l[r];
    }
}

extern "C" void kernel_launch(void* const* d_in, const int* in_sizes, int n_in,
                              void* d_out, int out_size, void* d_ws, size_t ws_size,
                              hipStream_t stream) {
  const float* X = (const float*)d_in[0];   // input_seq  [16,2048,256] f32
  const float* A = (const float*)d_in[1];   // attendee   [16,2048,256] f32
  const float* W = (const float*)d_in[2];   // W          [256,256]     f32
  float* out = (float*)d_out;

  char* ws = (char*)d_ws;
  bf16* WT = (bf16*)ws;                                        // 128 KB
  bf16* Qb = (bf16*)(ws + (size_t)D_ * D_ * 2);                // 16 MB
  bf16* Ab = (bf16*)(ws + (size_t)D_ * D_ * 2 +
                     (size_t)B_ * NI_ * D_ * 2);               // 16 MB

  wt_kernel<<<dim3(256), dim3(256), 0, stream>>>(W, WT);
  conv_kernel<<<dim3((B_ * NA_ * D_) / (256 * 8)), dim3(256), 0, stream>>>(A, Ab);
  qproj_kernel<<<dim3(256), dim3(256), 0, stream>>>(X, WT, Qb);

  const int SMEM = KT * KS * 2 + D_ * VS * 2 + 8 * 16 * PS * 2;  // 172032 B
  (void)hipFuncSetAttribute((const void*)flash_kernel,
                            hipFuncAttributeMaxDynamicSharedMemorySize, SMEM);
  flash_kernel<<<dim3(B_ * (NI_ / 128)), dim3(256), SMEM, stream>>>(Qb, Ab, out);
}
